// EdgeRelativeEmbedMLPPath_36507222016126
// MI455X (gfx1250) — matrix operand order repaired and checked by reading, not compiled, not run on hardware
//
#include <hip/hip_runtime.h>
#include <hip/hip_bf16.h>
#include <stddef.h>
#include <stdint.h>

typedef __bf16 bf16_t;
typedef __attribute__((ext_vector_type(16))) __bf16 v16bf;
typedef __attribute__((ext_vector_type(8)))  __bf16 v8bf;
typedef __attribute__((ext_vector_type(8)))  float  v8f;

#define NB    16
#define NN    8192
#define NE    8191
#define EPAD  8192
#define ADIM  256
#define VDIM  128

// Async global->LDS copy of 16B per lane (gfx1250 path, ASYNCcnt-tracked).
__device__ __forceinline__ void async_copy_b128(void* lds_dst, const void* gsrc) {
    unsigned ldsoff = (unsigned)(uintptr_t)lds_dst;   // flat-LDS aperture: low 32 bits = LDS byte offset
    asm volatile("global_load_async_to_lds_b128 %0, %1, off"
                 :: "v"(ldsoff), "v"(gsrc) : "memory");
}
__device__ __forceinline__ void wait_async_zero() {
    asm volatile("s_wait_asynccnt 0x0" ::: "memory");
}

// ---------------------------------------------------------------------------
// Precompute: fold W2 into the scalar conv kernel.
//   beff[(t*256+o)*256 + a] = sum_i K_a[o,i,t] * W2[i,a]      (bf16, B-matrix)
// ---------------------------------------------------------------------------
__global__ __launch_bounds__(256) void beff_kernel(const float* __restrict__ Ka,
                                                   const float* __restrict__ W2,
                                                   bf16_t* __restrict__ beff) {
    int to = blockIdx.x;                 // t*256 + o
    int t = to >> 8, o = to & 255;
    int a = threadIdx.x;
    float acc = 0.f;
    for (int i = 0; i < 256; ++i)
        acc += Ka[(o * 256 + i) * 4 + t] * W2[i * 256 + a];
    beff[(size_t)to * 256 + a] = (bf16_t)acc;
}

// bv[(t*128+o)*128 + i] = K_v[o,i,t]  (bf16 reformat for B fragments)
__global__ __launch_bounds__(256) void bv_kernel(const float* __restrict__ Kv,
                                                 bf16_t* __restrict__ bv) {
    int idx = blockIdx.x * 256 + threadIdx.x;      // 4*128*128 = 65536
    int t = idx >> 14;
    int o = (idx >> 7) & 127;
    int i = idx & 127;
    bv[idx] = (bf16_t)Kv[(o * 128 + i) * 4 + t];
}

// cb[o*4+t] = sum_i K_a[o,i,t] * b2[i]   (boundary-masked bias table)
__global__ __launch_bounds__(256) void cb_kernel(const float* __restrict__ Ka,
                                                 const float* __restrict__ b2,
                                                 float* __restrict__ cb) {
    int t = blockIdx.x;
    int o = threadIdx.x;
    float acc = 0.f;
    for (int i = 0; i < 256; ++i)
        acc += Ka[(o * 256 + i) * 4 + t] * b2[i];
    cb[o * 4 + t] = acc;
}

// ---------------------------------------------------------------------------
// Edge stage: vecs, norms, MLP layer 1 (+LeakyReLU) -> h (bf16),
// VecLinear -> y_v (bf16). 64 edges per 256-thread block.
// ---------------------------------------------------------------------------
__global__ __launch_bounds__(256) void edge_kernel(const float* __restrict__ pos0,
                                                   const float* __restrict__ pos1,
                                                   const float* __restrict__ Wv,
                                                   const float* __restrict__ W1,
                                                   const float* __restrict__ b1,
                                                   bf16_t* __restrict__ hbuf,
                                                   bf16_t* __restrict__ yvbuf) {
    __shared__ float s_norm[64][4];
    __shared__ float s_vec[64][12];            // [edge][i*3 + comp]
    __shared__ float sW1[ADIM * 4];
    __shared__ float sb1[ADIM];
    __shared__ float sWv[VDIM * 4];

    const int tid = threadIdx.x;
    const int b   = blockIdx.y;
    const int e0  = blockIdx.x * 64;

    for (int i = tid; i < ADIM * 4; i += 256) sW1[i] = W1[i];
    if (tid < ADIM) sb1[tid] = b1[tid];
    for (int i = tid; i < VDIM * 4; i += 256) sWv[i] = Wv[i];

    if (tid < 64) {
        int e = e0 + tid;
        float v[12];
        if (e < NE) {
            size_t p  = ((size_t)b * NN + e) * 3;
            float a0x = pos0[p],     a0y = pos0[p + 1], a0z = pos0[p + 2];
            float a1x = pos0[p + 3], a1y = pos0[p + 4], a1z = pos0[p + 5];
            float c0x = pos1[p],     c0y = pos1[p + 1], c0z = pos1[p + 2];
            float c1x = pos1[p + 3], c1y = pos1[p + 4], c1z = pos1[p + 5];
            v[0] = a1x - a0x; v[1]  = a1y - a0y; v[2]  = a1z - a0z;   // pos0[e+1]-pos0[e]
            v[3] = c1x - c0x; v[4]  = c1y - c0y; v[5]  = c1z - c0z;   // pos1[e+1]-pos1[e]
            v[6] = c0x - a1x; v[7]  = c0y - a1y; v[8]  = c0z - a1z;   // pos1[e]-pos0[e+1]
            v[9] = c1x - a0x; v[10] = c1y - a0y; v[11] = c1z - a0z;   // pos1[e+1]-pos0[e]
        } else {
            for (int i = 0; i < 12; ++i) v[i] = 0.f;
        }
        for (int i = 0; i < 12; ++i) s_vec[tid][i] = v[i];
        for (int f = 0; f < 4; ++f)
            s_norm[tid][f] = sqrtf(v[f*3]*v[f*3] + v[f*3+1]*v[f*3+1] + v[f*3+2]*v[f*3+2]);
    }
    __syncthreads();

    // h = LeakyReLU(W1 @ norms + b1): 64 edges x 256 outs, 8 bf16 per thread-iter
    for (int it = 0; it < 8; ++it) {
        int cidx = it * 256 + tid;          // 2048 chunks of 8
        int el = cidx >> 5;                 // 32 chunks per edge
        int a0 = (cidx & 31) * 8;
        int e  = e0 + el;
        if (e < NE) {
            float n0 = s_norm[el][0], n1 = s_norm[el][1];
            float n2 = s_norm[el][2], n3 = s_norm[el][3];
            v8bf out;
            #pragma unroll
            for (int j = 0; j < 8; ++j) {
                int a = a0 + j;
                float h = sb1[a] + sW1[a*4+0]*n0 + sW1[a*4+1]*n1
                                 + sW1[a*4+2]*n2 + sW1[a*4+3]*n3;
                h = (h >= 0.f) ? h : 0.2f * h;
                out[j] = (bf16_t)h;
            }
            *(v8bf*)&hbuf[((size_t)b * EPAD + e) * ADIM + a0] = out;
        }
    }

    // y_v = W_v @ vecs per component: 64 edges x 3 comps x 128 outs
    for (int it = 0; it < 12; ++it) {
        int cidx = it * 256 + tid;          // 3072 chunks of 8
        int el   = cidx / 48;               // 48 chunks per edge
        int rem  = cidx - el * 48;
        int comp = rem >> 4;
        int o0   = (rem & 15) * 8;
        int e    = e0 + el;
        if (e < NE) {
            float x0 = s_vec[el][0*3 + comp], x1 = s_vec[el][1*3 + comp];
            float x2 = s_vec[el][2*3 + comp], x3 = s_vec[el][3*3 + comp];
            v8bf out;
            #pragma unroll
            for (int j = 0; j < 8; ++j) {
                int o = o0 + j;
                out[j] = (bf16_t)(sWv[o*4+0]*x0 + sWv[o*4+1]*x1 +
                                  sWv[o*4+2]*x2 + sWv[o*4+3]*x3);
            }
            *(v8bf*)&yvbuf[(((size_t)b * 3 + comp) * EPAD + e) * VDIM + o0] = out;
        }
    }
}

// ---------------------------------------------------------------------------
// Fragment helpers (contiguous k order used here, A and B alike)
//  A 16x32: lane m=lane&15; half kb = (lane<16?0:16); k = kb+0..15 contiguous
//  B 32x16: lane n=lane&15; half       (lane<16?0:16); k = +0..15 contiguous
// ---------------------------------------------------------------------------
__device__ __forceinline__ v16bf load_a_frag(const bf16_t* p /* &lds[row][kk*32 + hi*16] */) {
    v8bf a0 = *(const v8bf*)p;
    v8bf a1 = *(const v8bf*)(p + 8);
    v16bf r;
    #pragma unroll
    for (int j = 0; j < 8; ++j) { r[j] = a0[j]; r[j + 8] = a1[j]; }
    return r;
}

// ---------------------------------------------------------------------------
// ScalConv as shift-accumulate GEMM:  a[n,o] = sum_t h[n-2+t,:] @ Keff_t[:,o]
// Block: 64 n-rows x 128 o-cols, 8 waves; wave w owns o-strip w, 4 n-tiles.
// A tile staged via async global->LDS (ASYNCcnt); OOB rows zero-filled (conv pad).
// ---------------------------------------------------------------------------
#define ASTR 264   // 256 + 8 bf16 pad (row stride, 16B aligned)

__global__ __launch_bounds__(256) void sconv_kernel(const bf16_t* __restrict__ hbuf,
                                                    const bf16_t* __restrict__ beff,
                                                    const float* __restrict__ cb,
                                                    const float* __restrict__ ba,
                                                    float* __restrict__ aout) {
    __shared__ bf16_t ldsA[67 * ASTR];         // rows r=0..66 <-> e = n0-2+r

    const int n0 = blockIdx.x * 64;
    const int o0 = blockIdx.y * 128;
    const int b  = blockIdx.z;

    // Stage A tile (h rows, zero-padded at tensor boundary)
    const bf16_t* hb = hbuf + (size_t)b * EPAD * ADIM;
    for (int idx = threadIdx.x; idx < 67 * 32; idx += 256) {
        int r  = idx >> 5;
        int c8 = (idx & 31) * 8;
        int e  = n0 - 2 + r;
        bf16_t* dst = &ldsA[r * ASTR + c8];
        if (e >= 0 && e < NE) {
            async_copy_b128(dst, &hb[(size_t)e * ADIM + c8]);
        } else {
            v8bf z;
            #pragma unroll
            for (int j = 0; j < 8; ++j) z[j] = (bf16_t)0.f;
            *(v8bf*)dst = z;
        }
    }
    wait_async_zero();
    __syncthreads();

    const int wave = threadIdx.x >> 5;
    const int lane = threadIdx.x & 31;
    const int m    = lane & 15;
    const int hi   = lane >> 4;
    const int orow = o0 + wave * 16 + m;

    v8f acc[4];
    #pragma unroll
    for (int q = 0; q < 4; ++q)
        #pragma unroll
        for (int r = 0; r < 8; ++r) acc[q][r] = 0.f;

    #pragma unroll
    for (int t = 0; t < 4; ++t) {
        const bf16_t* brow = beff + ((size_t)(t * 256) + orow) * ADIM;
        #pragma unroll
        for (int kk = 0; kk < 8; ++kk) {
            v16bf bf = *(const v16bf*)(brow + kk * 32 + hi * 16);   // L2-resident
            #pragma unroll
            for (int q = 0; q < 4; ++q) {
                int row = q * 16 + m + t;
                v16bf af = load_a_frag(&ldsA[row * ASTR + kk * 32 + hi * 16]);
                acc[q] = __builtin_amdgcn_wmma_f32_16x16x32_bf16(
                    false, af, false, bf, (short)0, acc[q], false, false);
            }
        }
    }

    // Epilogue: + b_a + boundary-masked folded b2 bias
    const float bias_o = ba[orow];
    float cb0 = cb[orow*4+0], cb1 = cb[orow*4+1], cb2 = cb[orow*4+2], cb3 = cb[orow*4+3];
    #pragma unroll
    for (int q = 0; q < 4; ++q) {
        #pragma unroll
        for (int r = 0; r < 8; ++r) {
            int M = r + hi * 8;
            int n = n0 + q * 16 + M;
            float bias = bias_o;
            if (n - 2 >= 0)      bias += cb0;
            if (n - 1 >= 0)      bias += cb1;
            if (n     <  NE)     bias += cb2;
            if (n + 1 <  NE)     bias += cb3;
            aout[((size_t)b * NN + n) * ADIM + orow] = acc[q][r] + bias;
        }
    }
}

// ---------------------------------------------------------------------------
// VecConv: v[n,o,k] = sum_t y_v[n-2+t,:,k] @ K_v[:, :, t]^T   (no bias)
// ---------------------------------------------------------------------------
#define VSTR 136   // 128 + 8 bf16 pad

__global__ __launch_bounds__(256) void vconv_kernel(const bf16_t* __restrict__ yvbuf,
                                                    const bf16_t* __restrict__ bv,
                                                    float* __restrict__ vout) {
    __shared__ bf16_t ldsA[67 * VSTR];

    const int n0   = blockIdx.x * 64;
    const int comp = blockIdx.y;
    const int b    = blockIdx.z;

    const bf16_t* yb = yvbuf + ((size_t)b * 3 + comp) * EPAD * VDIM;
    for (int idx = threadIdx.x; idx < 67 * 16; idx += 256) {
        int r  = idx >> 4;
        int c8 = (idx & 15) * 8;
        int e  = n0 - 2 + r;
        bf16_t* dst = &ldsA[r * VSTR + c8];
        if (e >= 0 && e < NE) {
            async_copy_b128(dst, &yb[(size_t)e * VDIM + c8]);
        } else {
            v8bf z;
            #pragma unroll
            for (int j = 0; j < 8; ++j) z[j] = (bf16_t)0.f;
            *(v8bf*)dst = z;
        }
    }
    wait_async_zero();
    __syncthreads();

    const int wave = threadIdx.x >> 5;
    const int lane = threadIdx.x & 31;
    const int m    = lane & 15;
    const int hi   = lane >> 4;
    const int orow = wave * 16 + m;         // o-tile covers all 128

    v8f acc[4];
    #pragma unroll
    for (int q = 0; q < 4; ++q)
        #pragma unroll
        for (int r = 0; r < 8; ++r) acc[q][r] = 0.f;

    #pragma unroll
    for (int t = 0; t < 4; ++t) {
        const bf16_t* brow = bv + ((size_t)(t * 128) + orow) * VDIM;
        #pragma unroll
        for (int kk = 0; kk < 4; ++kk) {
            v16bf bf = *(const v16bf*)(brow + kk * 32 + hi * 16);
            #pragma unroll
            for (int q = 0; q < 4; ++q) {
                int row = q * 16 + m + t;
                v16bf af = load_a_frag(&ldsA[row * VSTR + kk * 32 + hi * 16]);
                acc[q] = __builtin_amdgcn_wmma_f32_16x16x32_bf16(
                    false, af, false, bf, (short)0, acc[q], false, false);
            }
        }
    }

    #pragma unroll
    for (int q = 0; q < 4; ++q) {
        #pragma unroll
        for (int r = 0; r < 8; ++r) {
            int M = r + hi * 8;
            int n = n0 + q * 16 + M;
            vout[(((size_t)b * NN + n) * VDIM + orow) * 3 + comp] = acc[q][r];
        }
    }
}

// ---------------------------------------------------------------------------
extern "C" void kernel_launch(void* const* d_in, const int* in_sizes, int n_in,
                              void* d_out, int out_size, void* d_ws, size_t ws_size,
                              hipStream_t stream) {
    (void)in_sizes; (void)n_in; (void)out_size; (void)ws_size;
    const float* pos0 = (const float*)d_in[0];
    const float* pos1 = (const float*)d_in[1];
    const float* Wv   = (const float*)d_in[2];
    const float* W1   = (const float*)d_in[3];
    const float* b1   = (const float*)d_in[4];
    const float* W2   = (const float*)d_in[5];
    const float* b2   = (const float*)d_in[6];
    const float* Ka   = (const float*)d_in[7];
    const float* ba   = (const float*)d_in[8];
    const float* Kv   = (const float*)d_in[9];

    char* ws = (char*)d_ws;
    const size_t HB = (size_t)NB * EPAD * ADIM * 2;          // 67,108,864
    const size_t YB = (size_t)NB * 3 * EPAD * VDIM * 2;      // 100,663,296
    const size_t BE = (size_t)4 * 256 * 256 * 2;             //    524,288
    const size_t BV = (size_t)4 * 128 * 128 * 2;             //    131,072
    bf16_t* hbuf = (bf16_t*)(ws);
    bf16_t* yv   = (bf16_t*)(ws + HB);
    bf16_t* beff = (bf16_t*)(ws + HB + YB);
    bf16_t* bv   = (bf16_t*)(ws + HB + YB + BE);
    float*  cb   = (float*) (ws + HB + YB + BE + BV);

    float* aout = (float*)d_out;
    float* vout = aout + (size_t)NB * NN * ADIM;

    beff_kernel<<<dim3(1024), dim3(256), 0, stream>>>(Ka, W2, beff);
    bv_kernel  <<<dim3(256),  dim3(256), 0, stream>>>(Kv, bv);
    cb_kernel  <<<dim3(4),    dim3(256), 0, stream>>>(Ka, b2, cb);
    edge_kernel<<<dim3(128, NB), dim3(256), 0, stream>>>(pos0, pos1, Wv, W1, b1, hbuf, yv);
    sconv_kernel<<<dim3(NN/64, 2, NB), dim3(256), 0, stream>>>(hbuf, beff, cb, ba, aout);
    vconv_kernel<<<dim3(NN/64, 3, NB), dim3(256), 0, stream>>>(yv, bv, vout);
}
